// SoftHistogram_80857054314616
// MI455X (gfx1250) — compile-verified, matrix operand order checked by reading, not run on hardware
//
#include <hip/hip_runtime.h>

#define BINS 64
#define PLANES 24                 // 8*3
#define HW_PIX (512*512)
#define BLOCKS_PER_PLANE 128
#define THREADS 256
#define WAVES (THREADS/32)
#define PIX_PER_BLOCK (HW_PIX / BLOCKS_PER_PLANE)   // 2048
#define PIX_PER_WAVE  (PIX_PER_BLOCK / WAVES)       // 256
#define TILES_PER_WAVE (PIX_PER_WAVE / 32)          // 8

typedef __attribute__((ext_vector_type(16))) _Float16 v16h;
typedef __attribute__((ext_vector_type(8)))  float    v8f;

// ---------------------------------------------------------------------------
// Kernel 1: zero the per-plane partial-histogram accumulator in workspace.
// ---------------------------------------------------------------------------
__global__ void zero_partial(float* __restrict__ p, int n) {
    int i = blockIdx.x * blockDim.x + threadIdx.x;
    if (i < n) p[i] = 0.0f;
}

// ---------------------------------------------------------------------------
// Kernel 2: main soft-histogram.
// Each wave processes 32 pixels per tile:
//   - 16-bin Gaussian window per pixel (16 v_exp_f32 instead of 64)
//   - per-pixel normalization (fast v_rcp_f32) folded in before f16 convert
//   - tile staged in LDS bin-major [bin][pixel] as f16, one shared base
//     address + immediate DS offsets for the window scatter
//   - reduced over pixels with v_wmma_f32_16x16x32_f16 (A = ones)
// ---------------------------------------------------------------------------
__global__ __launch_bounds__(THREADS)
void soft_hist_main(const float* __restrict__ x, float* __restrict__ partial) {
    // 8 waves x (64 bins x 32 pixels) f16 = 32 KB LDS
    __shared__ _Float16 tile[WAVES][BINS * 32];

    const int lane  = threadIdx.x & 31;
    const int wave  = threadIdx.x >> 5;
    const int plane = blockIdx.y;
    const long long base = (long long)plane * HW_PIX
                         + (long long)blockIdx.x * PIX_PER_BLOCK
                         + (long long)wave * PIX_PER_WAVE;
    const float* xp = x + base + lane;

    _Float16* buf = &tile[wave][0];
    float4*   zb  = (float4*)buf;                      // bulk zeroing
    const unsigned int* bu = (const unsigned int*)buf; // f16-pair fragment loads

    // A matrix: 16x32 of f16 ones -> every row of D = column sums of B
    union { v16h h; _Float16 e[16]; } ones;
#pragma unroll
    for (int i = 0; i < 16; ++i) ones.e[i] = (_Float16)1.0f;

    v8f acc[4];
#pragma unroll
    for (int nb = 0; nb < 4; ++nb) acc[nb] = (v8f){0,0,0,0,0,0,0,0};

    const int n   = lane & 15;          // B-matrix column (bin within block)
    const int kh8 = (lane >> 4) * 8;    // K-half offset in u32 units (pixels/2)

    for (int t = 0; t < TILES_PER_WAVE; ++t) {
        // ---- zero this wave's 4 KB tile buffer (8 x b128 per lane) ----
        float4* zp = zb + lane;
#pragma unroll
        for (int q = 0; q < 8; ++q)
            zp[32 * q] = make_float4(0.f, 0.f, 0.f, 0.f);
        asm volatile("" ::: "memory");

        // ---- per-pixel windowed Gaussian kernel ----
        float xv = xp[t * 32];
        // pull next tile's line toward the caches while we burn exps
        __builtin_prefetch(xp + (t + 1) * 32, 0, 0);

        xv = fminf(fmaxf(xv, 0.0f), 1.0f);
        int r  = __float2int_rn(xv * 63.0f);
        int i0 = r - 7;
        i0 = max(0, min(48, i0)) & ~1;          // even start, window [i0, i0+16)

        float e[16];
        float z = 0.0f;
#pragma unroll
        for (int u = 0; u < 16; ++u) {
            float c = (float)(i0 + u) * (1.0f / 63.0f);
            float d = (xv - c) * 50.0f;         // 1/sigma = 50
            e[u] = __expf(-0.5f * d * d);
            z += e[u];
        }
        float w = __builtin_amdgcn_rcpf(z + 1e-12f);   // single v_rcp_f32

        // scatter normalized window into bin-major LDS (column = this lane):
        // one base address VGPR, 16 stores with immediate offsets u*64
        _Float16* wp = buf + i0 * 32 + lane;
#pragma unroll
        for (int u = 0; u < 16; ++u)
            wp[u * 32] = (_Float16)(e[u] * w);

        // cross-lane LDS consumption next: drain DS counter explicitly
        asm volatile("s_wait_dscnt 0x0" ::: "memory");

        // ---- WMMA reduction: 4 bin-blocks of 16, B = 32 pixels x 16 bins ----
#pragma unroll
        for (int nb = 0; nb < 4; ++nb) {
            union { v16h h; unsigned int u32[8]; } bf;
#pragma unroll
            for (int v = 0; v < 8; ++v)
                // f16 pair (pixels kh+2v, kh+2v+1) of bin row nb*16+n
                bf.u32[v] = bu[(nb * 16 + n) * 16 + kh8 + v];

            acc[nb] = __builtin_amdgcn_wmma_f32_16x16x32_f16(
                false, ones.h, false, bf.h, (short)0, acc[nb], false, false);
        }
        asm volatile("" ::: "memory");
    }

    // All 16 rows of D are identical column sums; row M=0 lives in VGPR0 of
    // lanes 0..15 (N = lane). Flush per-wave partials to global accumulator.
    if (lane < 16) {
#pragma unroll
        for (int nb = 0; nb < 4; ++nb)
            atomicAdd(&partial[plane * BINS + nb * 16 + lane], acc[nb][0]);
    }
}

// ---------------------------------------------------------------------------
// Kernel 3: final per-image normalization. The 1/P mean factor cancels:
// out = partial / sum_bins(partial).
// ---------------------------------------------------------------------------
__global__ __launch_bounds__(BINS)
void normalize_hist(const float* __restrict__ partial, float* __restrict__ out) {
    __shared__ float s[BINS];
    const int b = blockIdx.x, t = threadIdx.x;
    s[t] = partial[b * BINS + t];
    __syncthreads();
    float sum = 0.0f;
#pragma unroll
    for (int j = 0; j < BINS; ++j) sum += s[j];
    out[b * BINS + t] = s[t] / (sum + 1e-12f);
}

extern "C" void kernel_launch(void* const* d_in, const int* in_sizes, int n_in,
                              void* d_out, int out_size, void* d_ws, size_t ws_size,
                              hipStream_t stream) {
    const float* x = (const float*)d_in[0];   // (8,3,512,512) f32
    // d_in[1] = centers: linspace(0,1,64); recomputed analytically in-kernel.
    float* out     = (float*)d_out;           // (8,3,64) f32
    float* partial = (float*)d_ws;            // 24*64 f32 accumulator

    zero_partial<<<(PLANES * BINS + 255) / 256, 256, 0, stream>>>(partial, PLANES * BINS);

    dim3 grid(BLOCKS_PER_PLANE, PLANES);
    soft_hist_main<<<grid, THREADS, 0, stream>>>(x, partial);

    normalize_hist<<<PLANES, BINS, 0, stream>>>(partial, out);
}
